// BLSTM_33423435498266
// MI455X (gfx1250) — matrix operand order repaired and checked by reading, not compiled, not run on hardware
//
#include <hip/hip_runtime.h>
#include <hip/hip_bf16.h>

// ---------------- problem constants ----------------
#define T_STEPS  1024
#define BATCH    64
#define HID      512
#define NWG      4        // HID / 128 columns per workgroup
#define NTHREADS 256      // 8 waves (wave32)
#define HPAD     8        // f16 padding per LDS row (16B) to spread banks
#define HSTRIDE  (HID + HPAD)

typedef __attribute__((ext_vector_type(16))) _Float16 v16h;
typedef __attribute__((ext_vector_type(8)))  float    v8f;

union Frag16 { uint4 q[2]; v16h v; };
union Frag32 { float f[8];  v8f  v; };

__device__ __forceinline__ float sigm(float x)      { return 1.0f / (1.0f + __expf(-x)); }
__device__ __forceinline__ float tanh_fast(float x) { return 2.0f * sigm(2.0f * x) - 1.0f; }

// ---------------- prep kernels ----------------
__global__ void k_binarize(const float* __restrict__ src, _Float16* __restrict__ dst, int n) {
    int i = blockIdx.x * blockDim.x + threadIdx.x;
    if (i < n) {
        float v = src[i];
        float s = (v > 0.0f) ? 1.0f : ((v < 0.0f) ? -1.0f : 0.0f);  // matches jnp.sign
        dst[i] = (_Float16)s;
    }
}

__global__ void k_bias(const float* __restrict__ a, const float* __restrict__ b,
                       float* __restrict__ o, int n) {
    int i = blockIdx.x * blockDim.x + threadIdx.x;
    if (i < n) o[i] = a[i] + b[i];
}

// init h double-buffer slot 0 (zeros or given initial state) and reset the step counter
__global__ void k_init(_Float16* __restrict__ hbuf, const _Float16* __restrict__ hinit,
                       unsigned int* __restrict__ counter, int n) {
    int i = blockIdx.x * blockDim.x + threadIdx.x;
    if (i == 0) *counter = 0u;
    if (i < n) hbuf[i] = hinit ? hinit[i] : (_Float16)0.0f;
}

// ---------------- persistent LSTM layer kernel ----------------
// Gate layout (reference jnp.split): pre columns [0:H)=i, [H:2H)=f, [2H:3H)=g, [3H:4H)=o.
// Each workgroup owns 128 h-columns; each wave owns 16 of them across all 4 gates.
template<int KIN, bool XF32, bool REV>
__global__ __launch_bounds__(NTHREADS)
void lstm_recur(const void*      __restrict__ xsrc,   // [T,B,KIN] f32 (XF32) or f16
                const _Float16*  __restrict__ Wih,    // binarized [4H, KIN]
                const _Float16*  __restrict__ Whh,    // binarized [4H, H]
                const float*     __restrict__ bsum,   // bih + bhh, [4H]
                const float*     __restrict__ cinit,  // [B,H] f32 or nullptr (zeros)
                _Float16*        __restrict__ hbuf,   // [2,B,H] f16 double buffer (slot0 pre-inited)
                _Float16*        __restrict__ yf16,   // [T,B,H] f16 out or nullptr
                float*           __restrict__ yf32,   // [T,B,2H] f32 out slice or nullptr
                _Float16*        __restrict__ saveH,  // [B,H] f16 final h or nullptr
                float*           __restrict__ saveC,  // [B,H] f32 final c or nullptr
                unsigned int*    __restrict__ counter)
{
    extern __shared__ char smem[];
    _Float16* sh = (_Float16*)smem;              // staged h_prev [BATCH][HSTRIDE]
    _Float16* sx = sh + BATCH * HSTRIDE;         // staged x_t    [BATCH][KIN+HPAD]
    constexpr int XSTRIDE = KIN + HPAD;

    const int tid  = threadIdx.x;
    const int wave = tid >> 5;
    const int lane = tid & 31;
    const int half = lane >> 4;                  // 0: lanes 0-15, 1: lanes 16-31
    const int n16  = lane & 15;
    const int col  = blockIdx.x * 128 + wave * 16 + n16;   // h-column this lane owns

    float bias[4];
    #pragma unroll
    for (int g = 0; g < 4; ++g) bias[g] = bsum[g * HID + col];

    // cell state in registers: element (mt, r) <-> batch row m = mt*16 + r + 8*half
    Frag32 cst[4];
    #pragma unroll
    for (int mt = 0; mt < 4; ++mt) {
        #pragma unroll
        for (int r = 0; r < 8; ++r) {
            const int m = mt * 16 + r + 8 * half;
            cst[mt].f[r] = cinit ? cinit[m * HID + col] : 0.0f;
        }
    }

    for (int s = 0; s < T_STEPS; ++s) {
        const int tt = REV ? (T_STEPS - 1 - s) : s;

        // ---- stage h_{t-1} into LDS (b128 copies, padded rows) ----
        {
            const _Float16* hsrc = hbuf + (size_t)(s & 1) * (BATCH * HID);
            for (int i = tid; i < BATCH * (HID / 8); i += NTHREADS) {
                const int row = i >> 6, c = i & 63;
                *(uint4*)(sh + row * HSTRIDE + c * 8) =
                    *(const uint4*)(hsrc + row * HID + c * 8);
            }
        }
        // ---- stage x_t into LDS (layer0: f32 -> f16 on the fly) ----
        if (XF32) {
            const float* xp = (const float*)xsrc + (size_t)tt * (BATCH * KIN);
            constexpr int CH = KIN / 4;
            for (int i = tid; i < BATCH * CH; i += NTHREADS) {
                const int row = i / CH, c = i - row * CH;
                const float4 v = *(const float4*)(xp + row * KIN + c * 4);
                union { _Float16 h[4]; uint2 u; } p;
                p.h[0] = (_Float16)v.x; p.h[1] = (_Float16)v.y;
                p.h[2] = (_Float16)v.z; p.h[3] = (_Float16)v.w;
                *(uint2*)(sx + row * XSTRIDE + c * 4) = p.u;
            }
        } else {
            const _Float16* xp = (const _Float16*)xsrc + (size_t)tt * (BATCH * KIN);
            constexpr int CH = KIN / 8;
            for (int i = tid; i < BATCH * CH; i += NTHREADS) {
                const int row = i / CH, c = i - row * CH;
                *(uint4*)(sx + row * XSTRIDE + c * 8) =
                    *(const uint4*)(xp + row * KIN + c * 8);
            }
        }
        __syncthreads();

        // ---- accumulators: [gate][mtile] 16x16 f32, pre-loaded with bias ----
        Frag32 acc[4][4];
        #pragma unroll
        for (int g = 0; g < 4; ++g)
            #pragma unroll
            for (int mt = 0; mt < 4; ++mt)
                #pragma unroll
                for (int r = 0; r < 8; ++r) acc[g][mt].f[r] = bias[g];

        // ---- recurrent part: G += h_prev @ Bhh^T  (K = HID) ----
        #pragma unroll 1
        for (int ks = 0; ks < HID / 32; ++ks) {
            Frag16 A[4];
            #pragma unroll
            for (int mt = 0; mt < 4; ++mt) {
                // A 16x32 f16 layout: lane holds row m=lane%16, two contiguous 8-elem
                // chunks at k = ks*32 + 8*half and +16 (ISA 7.12.2 table).
                const _Float16* p = sh + (mt * 16 + n16) * HSTRIDE + ks * 32 + 8 * half;
                A[mt].q[0] = *(const uint4*)p;
                A[mt].q[1] = *(const uint4*)(p + 16);
            }
            #pragma unroll
            for (int g = 0; g < 4; ++g) {
                // B 32x16 f16 layout: lane holds col n=lane%16, two 8-elem chunks at
                // k = ks*32 + 8*half and +16 (same order as A), from row-major W[4H][K].
                Frag16 Bf;
                const _Float16* wp = Whh + (size_t)(g * HID + col) * HID + ks * 32 + 8 * half;
                Bf.q[0] = *(const uint4*)wp;
                Bf.q[1] = *(const uint4*)(wp + 16);
                #pragma unroll
                for (int mt = 0; mt < 4; ++mt)
                    acc[g][mt].v = __builtin_amdgcn_wmma_f32_16x16x32_f16(
                        false, A[mt].v, false, Bf.v, (short)0, acc[g][mt].v, false, false);
            }
        }

        // ---- fused input part: G += x_t @ Bih^T  (K = KIN) ----
        #pragma unroll 1
        for (int ks = 0; ks < KIN / 32; ++ks) {
            Frag16 A[4];
            #pragma unroll
            for (int mt = 0; mt < 4; ++mt) {
                const _Float16* p = sx + (mt * 16 + n16) * XSTRIDE + ks * 32 + 8 * half;
                A[mt].q[0] = *(const uint4*)p;
                A[mt].q[1] = *(const uint4*)(p + 16);
            }
            #pragma unroll
            for (int g = 0; g < 4; ++g) {
                Frag16 Bf;
                const _Float16* wp = Wih + (size_t)(g * HID + col) * KIN + ks * 32 + 8 * half;
                Bf.q[0] = *(const uint4*)wp;
                Bf.q[1] = *(const uint4*)(wp + 16);
                #pragma unroll
                for (int mt = 0; mt < 4; ++mt)
                    acc[g][mt].v = __builtin_amdgcn_wmma_f32_16x16x32_f16(
                        false, A[mt].v, false, Bf.v, (short)0, acc[g][mt].v, false, false);
            }
        }

        // ---- gates -> cell/hidden update; write h and outputs ----
        _Float16* hdst = hbuf + (size_t)((s + 1) & 1) * (BATCH * HID);
        #pragma unroll
        for (int mt = 0; mt < 4; ++mt) {
            #pragma unroll
            for (int r = 0; r < 8; ++r) {
                const int m = mt * 16 + r + 8 * half;
                const float ig = sigm(acc[0][mt].f[r]);
                const float fg = sigm(acc[1][mt].f[r]);
                const float cg = tanh_fast(acc[2][mt].f[r]);
                const float og = sigm(acc[3][mt].f[r]);
                const float c  = fg * cst[mt].f[r] + ig * cg;
                cst[mt].f[r]   = c;
                const float h  = og * tanh_fast(c);
                hdst[m * HID + col] = (_Float16)h;
                if (yf16) yf16[((size_t)tt * BATCH + m) * HID + col] = (_Float16)h;
                if (yf32) yf32[((size_t)tt * BATCH + m) * (2 * HID) + col] = h;
                if (saveC && s == T_STEPS - 1) {
                    saveC[m * HID + col] = c;
                    saveH[m * HID + col] = (_Float16)h;
                }
            }
        }

        // ---- grid-wide barrier across NWG co-resident workgroups ----
        __syncthreads();
        if (tid == 0) {
            __threadfence();                      // release stores of this step
            atomicAdd(counter, 1u);
            const unsigned int target = (unsigned int)NWG * (unsigned int)(s + 1);
            while (atomicAdd(counter, 0u) < target) __builtin_amdgcn_s_sleep(2);
        }
        __syncthreads();
        __threadfence();                          // acquire: see peers' h writes
    }
}

// ---------------- host launcher ----------------
extern "C" void kernel_launch(void* const* d_in, const int* in_sizes, int n_in,
                              void* d_out, int out_size, void* d_ws, size_t ws_size,
                              hipStream_t stream)
{
    (void)in_sizes; (void)n_in; (void)out_size; (void)ws_size;

    const float* x      = (const float*)d_in[0];
    const float* Wih[4] = {(const float*)d_in[1], (const float*)d_in[5],
                           (const float*)d_in[9], (const float*)d_in[13]};
    const float* Whh[4] = {(const float*)d_in[2], (const float*)d_in[6],
                           (const float*)d_in[10], (const float*)d_in[14]};
    const float* bih[4] = {(const float*)d_in[3], (const float*)d_in[7],
                           (const float*)d_in[11], (const float*)d_in[15]};
    const float* bhh[4] = {(const float*)d_in[4], (const float*)d_in[8],
                           (const float*)d_in[12], (const float*)d_in[16]};
    const int KinL[4] = {256, 512, 256, 512};   // f0, f1, r0, r1

    // workspace carve-out
    char* ws = (char*)d_ws;
    size_t off = 0;
    auto carve = [&](size_t bytes) -> char* {
        char* p = ws + off; off += (bytes + 255) & ~(size_t)255; return p;
    };

    _Float16* y0 = (_Float16*)carve((size_t)T_STEPS * BATCH * HID * 2);  // layer0 outputs (reused by r0)
    _Float16 *bWih[4], *bWhh[4]; float* bs[4];
    for (int l = 0; l < 4; ++l) {
        bWih[l] = (_Float16*)carve((size_t)4 * HID * KinL[l] * 2);
        bWhh[l] = (_Float16*)carve((size_t)4 * HID * HID * 2);
        bs[l]   = (float*)carve((size_t)4 * HID * 4);
    }
    _Float16* hbuf = (_Float16*)carve((size_t)2 * BATCH * HID * 2);
    _Float16* hf0  = (_Float16*)carve((size_t)BATCH * HID * 2);
    _Float16* hf1  = (_Float16*)carve((size_t)BATCH * HID * 2);
    float*    cf0  = (float*)carve((size_t)BATCH * HID * 4);
    float*    cf1  = (float*)carve((size_t)BATCH * HID * 4);
    unsigned int* counters = (unsigned int*)carve(4 * sizeof(unsigned int));

    // binarize weights (sign -> +/-1 in f16) and fold biases
    for (int l = 0; l < 4; ++l) {
        const int nih = 4 * HID * KinL[l];
        const int nhh = 4 * HID * HID;
        k_binarize<<<(nih + 255) / 256, 256, 0, stream>>>(Wih[l], bWih[l], nih);
        k_binarize<<<(nhh + 255) / 256, 256, 0, stream>>>(Whh[l], bWhh[l], nhh);
        k_bias<<<(4 * HID + 255) / 256, 256, 0, stream>>>(bih[l], bhh[l], bs[l], 4 * HID);
    }

    float* out = (float*)d_out;
    const size_t smemH = (size_t)BATCH * HSTRIDE * 2;
    const size_t smem256 = smemH + (size_t)BATCH * (256 + HPAD) * 2;
    const size_t smem512 = smemH + (size_t)BATCH * (512 + HPAD) * 2;
    const int nInit = BATCH * HID;

    // f0: x -> y0, save final (h,c)
    k_init<<<(nInit + 255) / 256, 256, 0, stream>>>(hbuf, (const _Float16*)nullptr, counters + 0, nInit);
    lstm_recur<256, true, false><<<NWG, NTHREADS, smem256, stream>>>(
        x, bWih[0], bWhh[0], bs[0], nullptr, hbuf, y0, nullptr, hf0, cf0, counters + 0);

    // f1: y0 -> out[:, :, 0:H], save final (h,c)
    k_init<<<(nInit + 255) / 256, 256, 0, stream>>>(hbuf, (const _Float16*)nullptr, counters + 1, nInit);
    lstm_recur<512, false, false><<<NWG, NTHREADS, smem512, stream>>>(
        y0, bWih[1], bWhh[1], bs[1], nullptr, hbuf, nullptr, out, hf1, cf1, counters + 1);

    // r0 (reverse): x -> y0 (reused), starts from f0's final states (reference aliasing)
    k_init<<<(nInit + 255) / 256, 256, 0, stream>>>(hbuf, hf0, counters + 2, nInit);
    lstm_recur<256, true, true><<<NWG, NTHREADS, smem256, stream>>>(
        x, bWih[2], bWhh[2], bs[2], cf0, hbuf, y0, nullptr, nullptr, nullptr, counters + 2);

    // r1 (reverse): y0r -> out[:, :, H:2H], starts from f1's final states
    k_init<<<(nInit + 255) / 256, 256, 0, stream>>>(hbuf, hf1, counters + 3, nInit);
    lstm_recur<512, false, true><<<NWG, NTHREADS, smem512, stream>>>(
        y0, bWih[3], bWhh[3], bs[3], cf1, hbuf, nullptr, out + HID, nullptr, nullptr, counters + 3);
}
